// HearingLoss_25271587569933
// MI455X (gfx1250) — matrix operand order repaired and checked by reading, not compiled, not run on hardware
//
#include <hip/hip_runtime.h>

typedef __attribute__((ext_vector_type(16))) __bf16 v16bf;
typedef __attribute__((ext_vector_type(8)))  __bf16 v8bf;
typedef __attribute__((ext_vector_type(8)))  float  v8f;

#define FIR_WAVES 4
#define MAX_WIN   2560   // >= KTOT_max(2016) + 240 + 256

// ---------------------------------------------------------------------------
// Build bf16 Toeplitz WMMA B-fragments for filter h[C][L].
// Fragment layout: frag[((c*nchunks + chunk)*32 + lane)*16 + slot]
//   B[q][m] = h[m + P - q],  q = chunk*32 + kk(lane>>4, slot) (K-row, table order), m = lane&15 (N-col)
// ---------------------------------------------------------------------------
__global__ void build_frag_kernel(const float* __restrict__ h,
                                  __bf16* __restrict__ frag,
                                  int C, int L, int KTOT) {
  const int  nchunks = KTOT >> 5;
  const long total   = (long)C * nchunks * 512;
  long i = (long)blockIdx.x * blockDim.x + threadIdx.x;
  if (i >= total) return;
  const int s     = (int)(i & 15);
  const int lane  = (int)((i >> 4) & 31);
  long rest       = i >> 9;
  const int chunk = (int)(rest % nchunks);
  const int c     = (int)(rest / nchunks);
  const int P = KTOT - 16;
  const int q = chunk * 32 + ((s < 8) ? (8 * (lane >> 4) + s) : (16 + 8 * (lane >> 4) + (s - 8)));
  const int k = (lane & 15) + P - q;
  float v = (k >= 0 && k < L) ? h[(long)c * L + k] : 0.0f;
  frag[i] = (__bf16)v;
}

// ---------------------------------------------------------------------------
// Grouped causal FIR via block-Toeplitz WMMA GEMM.
// One wave produces TWO adjacent 256-sample output tiles of one channel, so
// each filter fragment (B operand) is loaded once and used by two WMMAs.
//   y[c, 512*pair + 256*t + 16*i + m] = D_t[i][m]
//   A_t[i][q'] = x[c, 512*pair + 256*t - P + 16*i + 32*chunk + q'] (LDS, bf16)
//   B = precomputed Toeplitz filter fragments                      (global)
// ---------------------------------------------------------------------------
__global__ __launch_bounds__(FIR_WAVES * 32, 1)
void fir_wmma_kernel(const float* __restrict__ xin,
                     const __bf16* __restrict__ frag,
                     float* __restrict__ yout,
                     int N, int C, int KTOT, int bcast) {
  __shared__ alignas(16) __bf16 lds[FIR_WAVES][MAX_WIN];
  const int lane = threadIdx.x & 31;
  const int wave = threadIdx.x >> 5;
  const int pairsPerChan = (N + 511) >> 9;
  const long job = (long)blockIdx.x * FIR_WAVES + wave;
  if (job >= (long)C * pairsPerChan) return;   // wave-uniform: EXEC stays full
  const int c    = (int)(job / pairsPerChan);
  const int pair = (int)(job % pairsPerChan);
  const int P       = KTOT - 16;
  const int nchunks = KTOT >> 5;
  const int WIN     = KTOT + 240 + 256;
  const long xbase  = (long)pair * 512 - P;
  const float* xc   = xin + (bcast ? 0 : (long)c * N);

  // Stage the shared input window into LDS as bf16 (zero-padded at edges).
  for (int i = lane; i < WIN; i += 32) {
    long idx = xbase + i;
    float v = (idx >= 0 && idx < N) ? xc[idx] : 0.0f;
    lds[wave][i] = (__bf16)v;
  }
  asm volatile("s_wait_dscnt 0" ::: "memory");  // wave-local LDS RAW ordering

  // A-operand per-lane addressing per ISA 16-bit A-matrix layout:
  // lane -> M = lane&15, K groups [koff,koff+8) and [koff+16,koff+24).
  const int mrow = lane & 15;
  const int koff = (lane < 16) ? 0 : 8;
  const __bf16* aw = &lds[wave][(mrow << 4) + koff];
  const __bf16* fb = frag + (((long)c * nchunks) << 9) + ((long)lane << 4);

  v8f acc0 = {0.f, 0.f, 0.f, 0.f, 0.f, 0.f, 0.f, 0.f};
  v8f acc1 = {0.f, 0.f, 0.f, 0.f, 0.f, 0.f, 0.f, 0.f};
  for (int q = 0; q < nchunks; ++q) {
    v16bf b = *(const v16bf*)(fb + ((long)q << 9)); // 32B contiguous per lane
    v8bf lo0 = *(const v8bf*)(aw + (q << 5));            // tile 0
    v8bf hi0 = *(const v8bf*)(aw + (q << 5) + 16);
    v8bf lo1 = *(const v8bf*)(aw + (q << 5) + 256);      // tile 1 (+256 samp)
    v8bf hi1 = *(const v8bf*)(aw + (q << 5) + 272);
    v16bf a0 = __builtin_shufflevector(lo0, hi0, 0, 1, 2, 3, 4, 5, 6, 7,
                                                 8, 9, 10, 11, 12, 13, 14, 15);
    v16bf a1 = __builtin_shufflevector(lo1, hi1, 0, 1, 2, 3, 4, 5, 6, 7,
                                                 8, 9, 10, 11, 12, 13, 14, 15);
    acc0 = __builtin_amdgcn_wmma_f32_16x16x32_bf16(
        false, a0, false, b, (short)0, acc0, false, false);
    acc1 = __builtin_amdgcn_wmma_f32_16x16x32_bf16(
        false, a1, false, b, (short)0, acc1, false, false);
  }

  // D layout: lane -> n = lane&15, VGPR v -> row i = v + 8*(lane>>4).
  const int ncol = lane & 15;
  const int half = lane >> 4;
  float* yc = yout + (long)c * N;
  const long base = (long)pair * 512;
#pragma unroll
  for (int v = 0; v < 8; ++v) {
    long pos0 = base + 16 * (v + 8 * half) + ncol;
    long pos1 = pos0 + 256;
    if (pos0 < N) yc[pos0] = acc0[v];
    if (pos1 < N) yc[pos1] = acc1[v];
  }
}

// ---------------------------------------------------------------------------
// Elementwise helpers
// ---------------------------------------------------------------------------
__global__ void delay_kernel(const float* __restrict__ in, float* __restrict__ out,
                             const int* __restrict__ delays, int N, int C) {
  long i = (long)blockIdx.x * blockDim.x + threadIdx.x;
  if (i >= (long)N * C) return;
  int c = (int)(i / N);
  int n = (int)(i % N);
  long idx = (long)n + delays[c];
  out[i] = (idx < N) ? in[(long)c * N + idx] : 0.0f;
}

__global__ void abs_kernel(const float* __restrict__ in, float* __restrict__ out, long total) {
  long i = (long)blockIdx.x * blockDim.x + threadIdx.x;
  if (i < total) out[i] = fabsf(in[i]);
}

__global__ void revabs_kernel(const float* __restrict__ in, float* __restrict__ out,
                              int N, int C) {
  long i = (long)blockIdx.x * blockDim.x + threadIdx.x;
  if (i >= (long)N * C) return;
  int c = (int)(i / N);
  int n = (int)(i % N);
  out[i] = fabsf(in[(long)c * N + (N - 1 - n)]);
}

__global__ void copy_kernel(const float* __restrict__ in, float* __restrict__ out, long total) {
  long i = (long)blockIdx.x * blockDim.x + threadIdx.x;
  if (i < total) out[i] = in[i];
}

// Loudness recruitment + channel recombination. env arrives time-reversed
// (output of the backward LPF pass) so we index it reversed here.
__global__ void recruit_sum_kernel(const float* __restrict__ pass_del,
                                   const float* __restrict__ env_rev,
                                   const float* __restrict__ ratios,
                                   const float* __restrict__ eqloud,
                                   float* __restrict__ out, int N, int C) {
  int n = blockIdx.x * blockDim.x + threadIdx.x;
  if (n >= N) return;
  float s = 0.0f;
  for (int c = 0; c < C; ++c) {
    float env   = env_rev[(long)c * N + (N - 1 - n)];
    float emax  = powf(10.0f, 0.05f * (eqloud[c] - 100.0f));  // equiv0dBSPL=100, ahr=0
    float ecl   = fminf(fmaxf(env, 1e-9f), emax);
    float enorm = fmaxf(ecl / emax, 1e-6f);
    float ex    = fmaxf(ratios[c], -5.0f) - 1.0f;
    float g     = fminf(powf(enorm, ex), 100.0f);
    s += pass_del[(long)c * N + n] * g;
  }
  out[n] = s * 0.3162277660168379f;  // 10^(-0.05*10)
}

// ---------------------------------------------------------------------------
extern "C" void kernel_launch(void* const* d_in, const int* in_sizes, int n_in,
                              void* d_out, int out_size, void* d_ws, size_t ws_size,
                              hipStream_t stream) {
  const float* x        = (const float*)d_in[0];
  const float* src_fwd  = (const float*)d_in[1];
  const float* src_bwd  = (const float*)d_in[2];
  const float* lpf      = (const float*)d_in[3];
  const float* gtn_imp  = (const float*)d_in[4];
  const float* hp_imp   = (const float*)d_in[5];
  const float* env_imp  = (const float*)d_in[6];
  const float* ratios   = (const float*)d_in[7];
  const float* eqloud   = (const float*)d_in[8];
  const int*   delays   = (const int*)d_in[9];
  float* out = (float*)d_out;

  const int N = in_sizes[0];
  const int C = 28, S = 4;                 // S = START2POLE - 1
  const int Lsrc = in_sizes[1];            // 256
  const int Llpf = in_sizes[3];            // 133
  const int Lgt  = in_sizes[4] / C;        // 1000
  const int Lenv = in_sizes[6] / C;        // 2000

  auto ktot = [](int L) { return ((L + 15 + 31) / 32) * 32; };
  const int Ksrc = ktot(Lsrc), Klpf = ktot(Llpf), Kgt = ktot(Lgt), Kenv = ktot(Lenv);

  // -------- workspace layout --------
  char* ws = (char*)d_ws;
  size_t off = 0;
  auto alloc = [&](size_t bytes) -> char* {
    char* p = ws + off;
    off += (bytes + 255) & ~(size_t)255;
    return p;
  };
  auto fragBytes = [&](int c, int K) { return (size_t)c * (K >> 5) * 512 * 2; };

  __bf16* fragSF  = (__bf16*)alloc(fragBytes(1, Ksrc));
  __bf16* fragSB  = (__bf16*)alloc(fragBytes(1, Ksrc));
  __bf16* fragLP  = (__bf16*)alloc(fragBytes(1, Klpf));
  __bf16* fragGT  = (__bf16*)alloc(fragBytes(C, Kgt));
  __bf16* fragHP  = (__bf16*)alloc(fragBytes(C - S, Kgt));
  __bf16* fragENV = (__bf16*)alloc(fragBytes(C, Kenv));
  float* bufS1 = (float*)alloc((size_t)N * 4);
  float* bufS2 = (float*)alloc((size_t)N * 4);
  float* out1  = (float*)alloc((size_t)N * 4);
  float* bufA  = (float*)alloc((size_t)C * N * 4);
  float* bufB  = (float*)alloc((size_t)C * N * 4);
  float* bufC  = (float*)alloc((size_t)C * N * 4);

  // -------- launch helpers --------
  auto build = [&](const float* h, __bf16* frag, int c, int L, int K) {
    long total = (long)c * (K >> 5) * 512;
    build_frag_kernel<<<(int)((total + 255) / 256), 256, 0, stream>>>(h, frag, c, L, K);
  };
  auto fir = [&](const float* xin, const __bf16* frag, float* yout,
                 int c, int K, int bcast) {
    int pairs = (N + 511) >> 9;
    long jobs = (long)c * pairs;
    fir_wmma_kernel<<<(int)((jobs + FIR_WAVES - 1) / FIR_WAVES),
                      FIR_WAVES * 32, 0, stream>>>(xin, frag, yout, N, c, K, bcast);
  };
  const long CN = (long)C * N;
  const int EB = 256;
  auto eblocks = [&](long t) { return (int)((t + EB - 1) / EB); };

  // -------- build Toeplitz filter fragments --------
  build(src_fwd, fragSF, 1, Lsrc, Ksrc);
  build(src_bwd, fragSB, 1, Lsrc, Ksrc);
  build(lpf,     fragLP, 1, Llpf, Klpf);
  build(gtn_imp, fragGT, C, Lgt, Kgt);
  build(hp_imp,  fragHP, C - S, Lgt, Kgt);
  build(env_imp, fragENV, C, Lenv, Kenv);

  // -------- pipeline --------
  // free-field -> cochlea correction
  fir(x, fragSF, bufS1, 1, Ksrc, 0);
  // 4 cascaded gammatone passes (pass 1 broadcasts xc to all channels)
  fir(bufS1, fragGT, bufA, C, Kgt, 1);
  fir(bufA,  fragGT, bufB, C, Kgt, 0);
  fir(bufB,  fragGT, bufA, C, Kgt, 0);
  fir(bufA,  fragGT, bufB, C, Kgt, 0);
  // per-channel delay compensation
  delay_kernel<<<eblocks(CN), EB, 0, stream>>>(bufB, bufA, delays, N, C);
  // high-pass correction on channels S..C-1; pass-through channels 0..S-1
  copy_kernel<<<eblocks((long)S * N), EB, 0, stream>>>(bufA, bufB, (long)S * N);
  fir(bufA + (long)S * N, fragHP, bufB + (long)S * N, C - S, Kgt, 0);
  // envelope: forward LPF of |pass_del|, then backward LPF
  abs_kernel<<<eblocks(CN), EB, 0, stream>>>(bufB, bufA, CN);
  fir(bufA, fragENV, bufC, C, Kenv, 0);                    // env_f
  revabs_kernel<<<eblocks(CN), EB, 0, stream>>>(bufC, bufA, N, C);
  fir(bufA, fragENV, bufC, C, Kenv, 0);                    // env (time-reversed)
  // loudness recruitment + recombination
  recruit_sum_kernel<<<eblocks(N), EB, 0, stream>>>(bufB, bufC, ratios, eqloud,
                                                    out1, N, C);
  // cochlea -> free-field correction, then 18 kHz anti-alias LPF
  fir(out1, fragSB, bufS2, 1, Ksrc, 0);
  fir(bufS2, fragLP, out, 1, Klpf, 0);
}
